// SelfAttentionWithRoPE_5308579578508
// MI455X (gfx1250) — matrix operand order repaired and checked by reading, compiled, not run on hardware
//
#include <hip/hip_runtime.h>
#include <hip/hip_fp16.h>

typedef __attribute__((ext_vector_type(16))) _Float16 v16h;
typedef __attribute__((ext_vector_type(8)))  _Float16 v8h;
typedef __attribute__((ext_vector_type(8)))  float    v8f;

#define HIDDEN 2048
#define NHEADS 16
#define HDIM   128
#define SEQ    2048
#define BATCH  2
#define TOKENS (BATCH*SEQ)

__device__ __forceinline__ v16h concat8(v8h lo, v8h hi) {
    return __builtin_shufflevector(lo, hi, 0,1,2,3,4,5,6,7,8,9,10,11,12,13,14,15);
}
__device__ __forceinline__ v8f zero8f() {
    v8f z = {0.f,0.f,0.f,0.f,0.f,0.f,0.f,0.f};
    return z;
}
__device__ __forceinline__ v8f wmma_f16(v16h a, v16h b, v8f c) {
    return __builtin_amdgcn_wmma_f32_16x16x32_f16(false, a, false, b, (short)0, c, false, false);
}

// ---------------------------------------------------------------------------
// Prep: f32 -> f16 bulk convert (n must be a multiple of 2048)
// ---------------------------------------------------------------------------
__global__ __launch_bounds__(256)
void cvt_f16_kernel(const float* __restrict__ in, _Float16* __restrict__ out)
{
    size_t i = ((size_t)blockIdx.x * 256 + threadIdx.x) * 8;
    #pragma unroll
    for (int j = 0; j < 8; ++j) out[i + j] = (_Float16)in[i + j];
}

// ---------------------------------------------------------------------------
// Prep: RoPE tables  cos/sin[pos][j], pos<SEQ, j<HDIM/2
// ---------------------------------------------------------------------------
__global__ __launch_bounds__(256)
void rope_table_kernel(float* __restrict__ cosT, float* __restrict__ sinT)
{
    int id  = blockIdx.x * 256 + threadIdx.x;   // SEQ*64 threads
    int pos = id >> 6;
    int j   = id & 63;
    float freq = __powf(10000.f, -(float)(2 * j) * (1.f / HDIM));
    float ang  = (float)pos * freq;
    float sn, cs;
    __sincosf(ang, &sn, &cs);
    cosT[id] = cs;
    sinT[id] = sn;
}

// ---------------------------------------------------------------------------
// LDS-free GEMM: Y = A(f16) @ W(f16)^T + bias.  M=TOKENS, N=HIDDEN, K=HIDDEN.
// Fragments loaded straight from global (L2-resident f16 operands):
//   A frag : lane row fixed, two contiguous 16B chunks      (A 16x32 layout)
//   B frag : lane col (row of W) fixed, one contiguous 32B  (B 32x16 layout)
// Block = 8 waves as 2(M)x4(N); wave tile 64x64 -> 16 wmma / 16 b128-loads
// per 32-wide K step.
// MODE 0: Q -> RoPE(tables) * 1/sqrt(128) -> Qh[b][h][s][d]
// MODE 1: K -> RoPE(tables)               -> Kh[b][h][s][d]
// MODE 2: V -> transposed                 -> Vt[b][h][d][s]
// MODE 3: out-proj -> f32 d_out
// ---------------------------------------------------------------------------
template<int MODE>
__global__ __launch_bounds__(256)
void gemm_nt(const _Float16* __restrict__ A, const _Float16* __restrict__ Wh,
             const float* __restrict__ bias,
             const float* __restrict__ cosT, const float* __restrict__ sinT,
             _Float16* __restrict__ outH, float* __restrict__ outF)
{
    const int m0   = blockIdx.y * 128;
    const int n0   = blockIdx.x * 256;
    const int tid  = threadIdx.x;
    const int lane = tid & 31;
    const int wave = tid >> 5;
    const int wm   = wave >> 2;          // 0..1 : 64-row group
    const int wn   = wave & 3;           // 0..3 : 64-col group
    const int row16 = lane & 15;
    const int hlf   = lane >> 4;

    // per-lane base pointers (include the lane-half k offset)
    const _Float16* ap[4];
    const _Float16* bp[4];
    #pragma unroll
    for (int mt = 0; mt < 4; ++mt)
        ap[mt] = A + (size_t)(m0 + wm * 64 + mt * 16 + row16) * HIDDEN + 16 * hlf;
    #pragma unroll
    for (int nt = 0; nt < 4; ++nt)
        bp[nt] = Wh + (size_t)(n0 + wn * 64 + nt * 16 + row16) * HIDDEN + 16 * hlf;

    v8f acc[4][4];
    #pragma unroll
    for (int i = 0; i < 4; ++i)
        #pragma unroll
        for (int j = 0; j < 4; ++j) acc[i][j] = zero8f();

    for (int kk = 0; kk < HIDDEN; kk += 32) {
        v16h af[4], bf[4];
        #pragma unroll
        for (int mt = 0; mt < 4; ++mt) {
            const v8h* lo = (const v8h*)(ap[mt] + kk);        // k = kk+16h .. +7
            const v8h* hi = (const v8h*)(ap[mt] + kk + 8);    // k = kk+16h+8 ..
            af[mt] = concat8(lo[0], hi[0]);
        }
        #pragma unroll
        for (int nt = 0; nt < 4; ++nt) {
            const v8h* p = (const v8h*)(bp[nt] + kk);         // k = kk+16h .. +15
            bf[nt] = concat8(p[0], p[1]);
        }
        #pragma unroll
        for (int mt = 0; mt < 4; ++mt)
            #pragma unroll
            for (int nt = 0; nt < 4; ++nt)
                acc[mt][nt] = wmma_f16(af[mt], bf[nt], acc[mt][nt]);
    }

    // epilogue: C layout -> lane owns one column (N=row16), rows r + 8*hlf
    #pragma unroll
    for (int mt = 0; mt < 4; ++mt) {
        #pragma unroll
        for (int nt = 0; nt < 4; ++nt) {
            int ncol = n0 + wn * 64 + nt * 16 + row16;
            float bv = bias[ncol];
            #pragma unroll
            for (int r = 0; r < 8; ++r) {
                int mrow = m0 + wm * 64 + mt * 16 + r + 8 * hlf;
                float val = acc[mt][nt][r] + bv;
                if (MODE == 3) {
                    outF[(size_t)mrow * HIDDEN + ncol] = val;
                } else {
                    int b   = mrow >> 11;            // / SEQ
                    int pos = mrow & (SEQ - 1);
                    int h   = ncol >> 7;             // / HDIM
                    int d   = ncol & (HDIM - 1);
                    if (MODE == 2) {
                        outH[(((size_t)(b * NHEADS + h)) * HDIM + d) * SEQ + pos] =
                            (_Float16)val;
                    } else {
                        // RoPE via tables; partner element d^1 lives in lane^1
                        float partner = __shfl_xor(val, 1, 32);
                        int ti = pos * 64 + (d >> 1);
                        float cs = cosT[ti];
                        float sn = sinT[ti];
                        float o = (d & 1) ? (partner * sn + val * cs)
                                          : (val * cs - partner * sn);
                        if (MODE == 0) o *= 0.08838834764831845f; // 1/sqrt(128)
                        outH[(((size_t)(b * NHEADS + h)) * SEQ + pos) * HDIM + d] =
                            (_Float16)o;
                    }
                }
            }
        }
    }
}

// ---------------------------------------------------------------------------
// Flash attention: grid (B*NH, SEQ/128); 8 waves, each owns 16 q-rows x d=128.
// ---------------------------------------------------------------------------
__global__ __launch_bounds__(256)
void attention_kernel(const _Float16* __restrict__ Qh,
                      const _Float16* __restrict__ Kh,
                      const _Float16* __restrict__ Vt,
                      _Float16* __restrict__ Ctx)
{
    __shared__ _Float16 Pl[8 * 16 * 40];   // per-wave 16x32 P tile (stride 40)

    const int bh   = blockIdx.x;           // 0..31
    const int b    = bh >> 4, h = bh & 15;
    const int tid  = threadIdx.x;
    const int lane = tid & 31;
    const int wave = tid >> 5;
    const int row16 = lane & 15;
    const int hlf   = lane >> 4;

    const int qrow = blockIdx.y * 128 + wave * 16;
    const size_t qkBase = (size_t)bh * SEQ * HDIM;

    v16h qfrag[4];
    {
        const _Float16* qp = Qh + qkBase + (size_t)(qrow + row16) * HDIM;
        #pragma unroll
        for (int f = 0; f < 4; ++f) {
            const v8h* lo = (const v8h*)(qp + f * 32 + 16 * hlf);
            const v8h* hi = (const v8h*)(qp + f * 32 + 8 + 16 * hlf);
            qfrag[f] = concat8(lo[0], hi[0]);
        }
    }

    v8f oacc[8];
    #pragma unroll
    for (int t = 0; t < 8; ++t) oacc[t] = zero8f();
    float mstate[8], lstate[8];
    #pragma unroll
    for (int r = 0; r < 8; ++r) { mstate[r] = -1e30f; lstate[r] = 0.f; }

    _Float16* myP = &Pl[wave * 16 * 40];

    for (int kb = 0; kb < SEQ; kb += 32) {
        v8f sc[2];
        #pragma unroll
        for (int t2 = 0; t2 < 2; ++t2) {
            v8f s = zero8f();
            const _Float16* kp = Kh + qkBase + (size_t)(kb + t2 * 16 + row16) * HDIM;
            #pragma unroll
            for (int f = 0; f < 4; ++f) {
                const v8h* p = (const v8h*)(kp + f * 32 + 16 * hlf);
                v16h kf = concat8(p[0], p[1]);
                s = wmma_f16(qfrag[f], kf, s);
            }
            sc[t2] = s;
        }

        float mnew[8], pscale[8];
        #pragma unroll
        for (int r = 0; r < 8; ++r) {
            float mx = fmaxf(sc[0][r], sc[1][r]);
            #pragma unroll
            for (int m = 1; m < 16; m <<= 1) mx = fmaxf(mx, __shfl_xor(mx, m, 32));
            mnew[r]   = fmaxf(mstate[r], mx);
            pscale[r] = __expf(mstate[r] - mnew[r]);
        }
        #pragma unroll
        for (int r = 0; r < 8; ++r) {
            float p0 = __expf(sc[0][r] - mnew[r]);
            float p1 = __expf(sc[1][r] - mnew[r]);
            myP[(r + 8 * hlf) * 40 + row16]      = (_Float16)p0;
            myP[(r + 8 * hlf) * 40 + 16 + row16] = (_Float16)p1;
            float rs = p0 + p1;
            #pragma unroll
            for (int m = 1; m < 16; m <<= 1) rs += __shfl_xor(rs, m, 32);
            lstate[r] = lstate[r] * pscale[r] + rs;
            mstate[r] = mnew[r];
        }
        #pragma unroll
        for (int t = 0; t < 8; ++t)
            #pragma unroll
            for (int r = 0; r < 8; ++r)
                oacc[t][r] = oacc[t][r] * pscale[r];

        v16h pfrag;
        {
            const v8h* lo = (const v8h*)&myP[row16 * 40 + 16 * hlf];
            const v8h* hi = (const v8h*)&myP[row16 * 40 + 8 + 16 * hlf];
            pfrag = concat8(lo[0], hi[0]);
        }

        const _Float16* vp = Vt + (size_t)bh * HDIM * SEQ;
        #pragma unroll
        for (int t = 0; t < 8; ++t) {
            const v8h* p = (const v8h*)(vp + (size_t)(t * 16 + row16) * SEQ
                                           + kb + 16 * hlf);
            v16h vf = concat8(p[0], p[1]);
            oacc[t] = wmma_f16(pfrag, vf, oacc[t]);
        }
    }

    const size_t ctxRow0 = ((size_t)b * SEQ + qrow) * HIDDEN + (size_t)h * HDIM;
    #pragma unroll
    for (int r = 0; r < 8; ++r) {
        float inv = 1.f / lstate[r];
        int soff = r + 8 * hlf;
        #pragma unroll
        for (int t = 0; t < 8; ++t) {
            Ctx[ctxRow0 + (size_t)soff * HIDDEN + t * 16 + row16] =
                (_Float16)(oacc[t][r] * inv);
        }
    }
}

// ---------------------------------------------------------------------------
extern "C" void kernel_launch(void* const* d_in, const int* in_sizes, int n_in,
                              void* d_out, int out_size, void* d_ws, size_t ws_size,
                              hipStream_t stream) {
    const float* X  = (const float*)d_in[0];
    const float* Wq = (const float*)d_in[1];
    const float* bq = (const float*)d_in[2];
    const float* Wk = (const float*)d_in[3];
    const float* bk = (const float*)d_in[4];
    const float* Wv = (const float*)d_in[5];
    const float* bv = (const float*)d_in[6];
    const float* Wo = (const float*)d_in[7];
    const float* bo = (const float*)d_in[8];
    float* out = (float*)d_out;

    // workspace layout (stream ordering makes the reuse safe):
    //   [0,16M)     Xh  (f16 hidden)   -- reused as Ctx after QKV gemms
    //   [16M,24M)   Wh  (f16 weight, rotating: Wq -> Wk -> Wv -> Wo)
    //   [24M,40M)   Qh
    //   [40M,56M)   Kh
    //   [56M,72M)   Vt  (transposed V)
    //   [72M,72.5M) cosT   [72.5M,73M) sinT
    char* ws = (char*)d_ws;
    _Float16* Xh   = (_Float16*)(ws);
    _Float16* Wh   = (_Float16*)(ws + ((size_t)16 << 20));
    _Float16* Qh   = (_Float16*)(ws + ((size_t)24 << 20));
    _Float16* Kh   = (_Float16*)(ws + ((size_t)40 << 20));
    _Float16* Vt   = (_Float16*)(ws + ((size_t)56 << 20));
    float*    cosT = (float*)   (ws + ((size_t)72 << 20));
    float*    sinT = (float*)   (ws + ((size_t)72 << 20) + ((size_t)512 << 10));
    _Float16* Ctx  = Xh;   // reuse: Xh dead after V projection

    dim3 blk(256);
    dim3 gGemm(HIDDEN / 256, TOKENS / 128);       // (8, 32)
    dim3 gAtt(BATCH * NHEADS, SEQ / 128);         // (32, 16)
    const int nbX = (TOKENS * HIDDEN) / 2048;     // 4096 blocks
    const int nbW = (HIDDEN * HIDDEN) / 2048;     // 2048 blocks

    rope_table_kernel<<<(SEQ * 64) / 256, blk, 0, stream>>>(cosT, sinT);
    cvt_f16_kernel<<<nbX, blk, 0, stream>>>(X, Xh);

    cvt_f16_kernel<<<nbW, blk, 0, stream>>>(Wq, Wh);
    gemm_nt<0><<<gGemm, blk, 0, stream>>>(Xh, Wh, bq, cosT, sinT, Qh, nullptr);

    cvt_f16_kernel<<<nbW, blk, 0, stream>>>(Wk, Wh);
    gemm_nt<1><<<gGemm, blk, 0, stream>>>(Xh, Wh, bk, cosT, sinT, Kh, nullptr);

    cvt_f16_kernel<<<nbW, blk, 0, stream>>>(Wv, Wh);
    gemm_nt<2><<<gGemm, blk, 0, stream>>>(Xh, Wh, bv, cosT, sinT, Vt, nullptr);

    attention_kernel<<<gAtt, blk, 0, stream>>>(Qh, Kh, Vt, Ctx);

    cvt_f16_kernel<<<nbW, blk, 0, stream>>>(Wo, Wh);
    gemm_nt<3><<<gGemm, blk, 0, stream>>>(Ctx, Wh, bo, cosT, sinT, nullptr, out);
}
